// MoE_8770323218863
// MI455X (gfx1250) — compile-verified, matrix operand order checked by reading, not run on hardware
//
#include <hip/hip_runtime.h>
#include <hip/hip_bf16.h>
#include <math.h>

// Problem constants (B=4, S=2048, H=1024, F=4096, E=8, top-2, CF=1.25)
#define E_    8
#define NTOK  8192
#define H_    1024
#define F_    4096
#define CAP_  1280          // ceil(8192/8 * 1.25)
#define EPS_  1e-5f

typedef __attribute__((ext_vector_type(16))) __bf16 v16bf;
typedef __attribute__((ext_vector_type(2)))  __bf16 v2bf;
typedef __attribute__((ext_vector_type(8)))  float  v8f;
typedef __attribute__((ext_vector_type(4)))  int    v4i;

union Frag {                 // 16 bf16 = 8 VGPRs, built from two 16B LDS reads
    v16bf v;
    uint4 q[2];
};

#if defined(__gfx1250__) && __has_builtin(__builtin_amdgcn_global_load_async_to_lds_b128)
#define ASYNC_LDS 1
#else
#define ASYNC_LDS 0
#endif

__device__ __forceinline__ unsigned short f2bf(float x) {
    unsigned u = __float_as_uint(x);                     // RNE fp32 -> bf16
    return (unsigned short)((u + 0x7FFFu + ((u >> 16) & 1u)) >> 16);
}
// Pack two fp32 into packed bf16x2 (one v_cvt_pk_bf16_f32 when available)
__device__ __forceinline__ unsigned pack2bf(float a, float b) {
#if __has_builtin(__builtin_amdgcn_cvt_pk_bf16_f32)
    union { v2bf v; unsigned u; } cv;
    cv.v = __builtin_amdgcn_cvt_pk_bf16_f32(a, b);
    return cv.u;
#else
    return (unsigned)f2bf(a) | ((unsigned)f2bf(b) << 16);
#endif
}
__device__ __forceinline__ float geluf(float x) {        // exact erf GELU
    return 0.5f * x * (1.0f + erff(x * 0.70710678118654752f));
}

#if ASYNC_LDS
__device__ __forceinline__ void async_copy16(const unsigned short* g, unsigned short* l) {
    __builtin_amdgcn_global_load_async_to_lds_b128((v4i*)g, (v4i*)l, 0, 0);
}
__device__ __forceinline__ void wait_async0() {
#if __has_builtin(__builtin_amdgcn_s_wait_asynccnt)
    __builtin_amdgcn_s_wait_asynccnt(0);
#else
    asm volatile("s_wait_asynccnt 0x0" ::: "memory");
#endif
}
#endif

// ---------------- init ----------------
__global__ void k_init_acc(const float* __restrict__ hidden,
                           float* __restrict__ acc, int n4) {
    int i = blockIdx.x * blockDim.x + threadIdx.x;
    if (i < n4) ((float4*)acc)[i] = ((const float4*)hidden)[i];
}
__global__ void k_init_route(int* __restrict__ tok, float* __restrict__ gate,
                             int* __restrict__ cnt) {
    int i = blockIdx.x * blockDim.x + threadIdx.x;
    if (i < E_ * CAP_) { tok[i] = 0; gate[i] = 0.0f; }
    if (i < E_) cnt[i] = 0;
}

// ---------------- router: one wave32 per token ----------------
__global__ void k_router(const float* __restrict__ x, const float* __restrict__ Wr,
                         const float* __restrict__ br, int* __restrict__ tok,
                         float* __restrict__ gate, int* __restrict__ cnt) {
    const int wave = threadIdx.x >> 5;
    const int lane = threadIdx.x & 31;
    const int token = blockIdx.x * 8 + wave;
    const float* xr = x + (size_t)token * H_;
    float p[E_];
#pragma unroll
    for (int e = 0; e < E_; ++e) p[e] = 0.0f;
    for (int h = lane; h < H_; h += 32) {
        float xv = xr[h];
        const float* wrow = Wr + h * E_;
#pragma unroll
        for (int e = 0; e < E_; ++e) p[e] += xv * wrow[e];
    }
#pragma unroll
    for (int off = 16; off >= 1; off >>= 1)
#pragma unroll
        for (int e = 0; e < E_; ++e) p[e] += __shfl_xor(p[e], off, 32);
    if (lane == 0) {
        float l[E_];
#pragma unroll
        for (int e = 0; e < E_; ++e) l[e] = p[e] + br[e];
        int e1 = 0; float v1 = l[0];
#pragma unroll
        for (int e = 1; e < E_; ++e) if (l[e] > v1) { v1 = l[e]; e1 = e; }
        int e2 = (e1 == 0) ? 1 : 0; float v2 = l[e2];
#pragma unroll
        for (int e = 0; e < E_; ++e)
            if (e != e1 && l[e] > v2) { v2 = l[e]; e2 = e; }
        float ex = __expf(v2 - v1);          // softmax over top-2
        float w1 = 1.0f / (1.0f + ex);
        float w2 = 1.0f - w1;
        int s1 = atomicAdd(&cnt[e1], 1);
        if (s1 < CAP_) { tok[e1 * CAP_ + s1] = token; gate[e1 * CAP_ + s1] = w1; }
        int s2 = atomicAdd(&cnt[e2], 1);
        if (s2 < CAP_) { tok[e2 * CAP_ + s2] = token; gate[e2 * CAP_ + s2] = w2; }
    }
}

// ---------------- gather + fp32->bf16 convert ----------------
__global__ void k_gather(const float* __restrict__ x, const int* __restrict__ tok,
                         unsigned short* __restrict__ Xe) {
    const int row = blockIdx.x;              // 0 .. E*CAP-1
    const int t = threadIdx.x;               // 256 threads, 4 floats each
    const int token = tok[row];
    float4 v = ((const float4*)(x + (size_t)token * H_))[t];
    uint2 o;
    o.x = pack2bf(v.x, v.y);
    o.y = pack2bf(v.z, v.w);
    ((uint2*)(Xe + (size_t)row * H_))[t] = o;
}

// ---------------- WMMA bf16 GEMM core (double buffered) ----------------
// Block: 256 thr (8 waves), tile 128(M) x 128(N), BK=32, ping-pong LDS.
// Wave grid 4(M) x 2(N): each wave 32x64 = 2x4 WMMA 16x16 accumulators.
// A tiles (bf16 in memory): async global->LDS copy (ASYNCcnt) when available.
// B tiles (fp32 weights): global->regs (overlapping WMMA), convert+transpose
// into LDS [N=128][K=40] so both fragments read as conflict-free b128.
// MODE 0: +bias, erf-GELU, store bf16 (GEMM1).
// MODE 1: +bias, *gate, atomic scatter-add into acc (GEMM2 + combine).
template <int MODE>
__global__ __launch_bounds__(256)
void k_gemm(const unsigned short* __restrict__ Abase,   // bf16 [E][Mtot][Ktot]
            const float* __restrict__ Bbase,            // fp32 [E][Ktot][Ntot]
            const float* __restrict__ bias,             // fp32 [E][Ntot]
            unsigned short* __restrict__ H1out,         // MODE 0 dest
            const int* __restrict__ tokA,
            const float* __restrict__ gateA,
            float* __restrict__ accOut,                 // MODE 1 dest
            int Mtot, int Ntot, int Ktot) {
    __shared__ __align__(16) unsigned short As[2][128 * 40];
    __shared__ __align__(16) unsigned short Bs[2][128 * 40];

    const int e  = blockIdx.z;
    const int mT = blockIdx.y * 128;
    const int nT = blockIdx.x * 128;
    const unsigned short* A = Abase + (size_t)e * Mtot * Ktot;
    const float* Bm = Bbase + (size_t)e * Ktot * Ntot;

    const int t = threadIdx.x;
    const int wave = t >> 5, lane = t & 31;
    const int wm = wave & 3, wn = wave >> 2;
    const int laneLo = lane & 15, laneHi = lane >> 4;
    const int nk = Ktot >> 5;

    // Per-thread tile-copy coordinates
    int arow[2], aoff[2];                   // A: 2 x 16B chunk per thread
#pragma unroll
    for (int c = 0; c < 2; ++c) {
        int chunk = t + c * 256;
        arow[c] = chunk >> 2;
        aoff[c] = (chunk & 3) * 8;          // bf16 units
    }
    int bn[4], bk0[4];                      // B: 4 items x 4 consecutive K
#pragma unroll
    for (int i = 0; i < 4; ++i) {
        int item = i * 256 + t;
        bn[i] = item & 127;
        bk0[i] = (item >> 7) * 4;
    }

    v8f accv[2][4];
    {
        v8f z = {};
#pragma unroll
        for (int mi = 0; mi < 2; ++mi)
#pragma unroll
            for (int ni = 0; ni < 4; ++ni) accv[mi][ni] = z;
    }

    float bregs[16];
#if !ASYNC_LDS
    uint4 aregs[2];
#endif

    // ---- prologue: fetch tile 0 into buffer 0 ----
#if ASYNC_LDS
#pragma unroll
    for (int c = 0; c < 2; ++c)
        async_copy16(A + (size_t)(mT + arow[c]) * Ktot + aoff[c],
                     &As[0][arow[c] * 40 + aoff[c]]);
#else
#pragma unroll
    for (int c = 0; c < 2; ++c)
        aregs[c] = *(const uint4*)(A + (size_t)(mT + arow[c]) * Ktot + aoff[c]);
#endif
#pragma unroll
    for (int i = 0; i < 4; ++i)
#pragma unroll
        for (int j = 0; j < 4; ++j)
            bregs[i * 4 + j] = Bm[(size_t)(bk0[i] + j) * Ntot + (nT + bn[i])];
#pragma unroll
    for (int i = 0; i < 4; ++i) {
        uint2 o;
        o.x = pack2bf(bregs[i * 4 + 0], bregs[i * 4 + 1]);
        o.y = pack2bf(bregs[i * 4 + 2], bregs[i * 4 + 3]);
        *(uint2*)(&Bs[0][bn[i] * 40 + bk0[i]]) = o;
    }
#if ASYNC_LDS
    wait_async0();
#else
#pragma unroll
    for (int c = 0; c < 2; ++c)
        *(uint4*)(&As[0][arow[c] * 40 + aoff[c]]) = aregs[c];
#endif
    __syncthreads();

    // ---- main loop: one barrier per K-step, next tile in flight ----
    for (int k = 0; k < nk; ++k) {
        const int cur = k & 1, nxt = cur ^ 1;
        const bool hasNext = (k + 1 < nk);
        const int ktn = (k + 1) << 5;

        if (hasNext) {                      // issue next tile's global traffic
#if ASYNC_LDS
#pragma unroll
            for (int c = 0; c < 2; ++c)
                async_copy16(A + (size_t)(mT + arow[c]) * Ktot + ktn + aoff[c],
                             &As[nxt][arow[c] * 40 + aoff[c]]);
#else
#pragma unroll
            for (int c = 0; c < 2; ++c)
                aregs[c] = *(const uint4*)(A + (size_t)(mT + arow[c]) * Ktot + ktn + aoff[c]);
#endif
#pragma unroll
            for (int i = 0; i < 4; ++i)
#pragma unroll
                for (int j = 0; j < 4; ++j)
                    bregs[i * 4 + j] = Bm[(size_t)(ktn + bk0[i] + j) * Ntot + (nT + bn[i])];
        }

        // compute current tile
        const unsigned short* Ac = As[cur];
        const unsigned short* Bc = Bs[cur];
        Frag fa[2], fb[4];
#pragma unroll
        for (int mi = 0; mi < 2; ++mi) {
            int row = wm * 32 + mi * 16 + laneLo;
            fa[mi].q[0] = *(const uint4*)(Ac + row * 40 + laneHi * 8);
            fa[mi].q[1] = *(const uint4*)(Ac + row * 40 + laneHi * 8 + 16);
        }
#pragma unroll
        for (int ni = 0; ni < 4; ++ni) {
            int col = wn * 64 + ni * 16 + laneLo;
            fb[ni].q[0] = *(const uint4*)(Bc + col * 40 + laneHi * 8);
            fb[ni].q[1] = *(const uint4*)(Bc + col * 40 + laneHi * 8 + 16);
        }
#pragma unroll
        for (int mi = 0; mi < 2; ++mi)
#pragma unroll
            for (int ni = 0; ni < 4; ++ni)
                accv[mi][ni] = __builtin_amdgcn_wmma_f32_16x16x32_bf16(
                    false, fa[mi].v, false, fb[ni].v,
                    (short)0, accv[mi][ni], false, false);

        if (hasNext) {                      // drain next tile into LDS[nxt]
#pragma unroll
            for (int i = 0; i < 4; ++i) {
                uint2 o;
                o.x = pack2bf(bregs[i * 4 + 0], bregs[i * 4 + 1]);
                o.y = pack2bf(bregs[i * 4 + 2], bregs[i * 4 + 3]);
                *(uint2*)(&Bs[nxt][bn[i] * 40 + bk0[i]]) = o;
            }
#if ASYNC_LDS
            wait_async0();
#else
#pragma unroll
            for (int c = 0; c < 2; ++c)
                *(uint4*)(&As[nxt][arow[c] * 40 + aoff[c]]) = aregs[c];
#endif
        }
        __syncthreads();
    }

    // Epilogue. C/D layout: lanes 0-15 -> N=lane, M = r (VGPR r);
    //                       lanes 16-31 -> N=lane-16, M = 8 + r.
#pragma unroll
    for (int mi = 0; mi < 2; ++mi) {
#pragma unroll
        for (int ni = 0; ni < 4; ++ni) {
            int n_g = nT + wn * 64 + ni * 16 + laneLo;
            float bv = bias[e * Ntot + n_g];
            int mrow = mT + wm * 32 + mi * 16 + laneHi * 8;
            if (MODE == 0) {
#pragma unroll
                for (int r = 0; r < 8; ++r) {
                    float v = accv[mi][ni][r] + bv;
                    H1out[(size_t)e * Mtot * Ntot + (size_t)(mrow + r) * Ntot + n_g] =
                        f2bf(geluf(v));
                }
            } else {
#pragma unroll
                for (int r = 0; r < 8; ++r) {
                    int m_g = mrow + r;
                    int token = tokA[e * CAP_ + m_g];
                    float g = gateA[e * CAP_ + m_g];
                    float v = (accv[mi][ni][r] + bv) * g;
                    atomicAdd(accOut + (size_t)token * H_ + n_g, v);
                }
            }
        }
    }
}

// ---------------- LayerNorm (per token row) ----------------
__global__ void k_layernorm(const float* __restrict__ acc,
                            const float* __restrict__ gamma,
                            const float* __restrict__ beta,
                            float* __restrict__ out) {
    __shared__ float s_sum[8], s_sq[8];
    const int row = blockIdx.x;
    const int t = threadIdx.x;
    float4 v = ((const float4*)(acc + (size_t)row * H_))[t];
    float s = v.x + v.y + v.z + v.w;
    float q = v.x * v.x + v.y * v.y + v.z * v.z + v.w * v.w;
#pragma unroll
    for (int off = 16; off >= 1; off >>= 1) {
        s += __shfl_xor(s, off, 32);
        q += __shfl_xor(q, off, 32);
    }
    const int wave = t >> 5, lane = t & 31;
    if (lane == 0) { s_sum[wave] = s; s_sq[wave] = q; }
    __syncthreads();
    float S = 0.0f, Q = 0.0f;
#pragma unroll
    for (int w = 0; w < 8; ++w) { S += s_sum[w]; Q += s_sq[w]; }
    float mu = S * (1.0f / H_);
    float var = Q * (1.0f / H_) - mu * mu;
    float rs = rsqrtf(var + EPS_);
    float4 g = ((const float4*)gamma)[t];
    float4 b = ((const float4*)beta)[t];
    float4 o;
    o.x = (v.x - mu) * rs * g.x + b.x;
    o.y = (v.y - mu) * rs * g.y + b.y;
    o.z = (v.z - mu) * rs * g.z + b.z;
    o.w = (v.w - mu) * rs * g.w + b.w;
    ((float4*)(out + (size_t)row * H_))[t] = o;
}

extern "C" void kernel_launch(void* const* d_in, const int* in_sizes, int n_in,
                              void* d_out, int out_size, void* d_ws, size_t ws_size,
                              hipStream_t stream) {
    (void)in_sizes; (void)n_in; (void)out_size; (void)ws_size;
    const float* x     = (const float*)d_in[0];   // hidden_states [4,2048,1024]
    const float* Wr    = (const float*)d_in[1];   // [H,E]
    const float* br    = (const float*)d_in[2];   // [E]
    const float* W1    = (const float*)d_in[3];   // [E,H,F]
    const float* b1    = (const float*)d_in[4];   // [E,F]
    const float* W2    = (const float*)d_in[5];   // [E,F,H]
    const float* b2    = (const float*)d_in[6];   // [E,H]
    const float* gamma = (const float*)d_in[7];   // [H]
    const float* beta  = (const float*)d_in[8];   // [H]
    float* out = (float*)d_out;

    char* ws = (char*)d_ws;
    size_t offXe   = 0;                                   // bf16 [E,CAP,H]
    size_t offH1   = offXe  + (size_t)E_ * CAP_ * H_ * 2; // bf16 [E,CAP,F]
    size_t offAcc  = offH1  + (size_t)E_ * CAP_ * F_ * 2; // f32  [N,H]
    size_t offTok  = offAcc + (size_t)NTOK * H_ * 4;      // i32  [E,CAP]
    size_t offGate = offTok + (size_t)E_ * CAP_ * 4;      // f32  [E,CAP]
    size_t offCnt  = offGate + (size_t)E_ * CAP_ * 4;     // i32  [E]
    unsigned short* Xe = (unsigned short*)(ws + offXe);
    unsigned short* H1 = (unsigned short*)(ws + offH1);
    float* acc  = (float*)(ws + offAcc);
    int* tokA   = (int*)(ws + offTok);
    float* gateA = (float*)(ws + offGate);
    int* cnt    = (int*)(ws + offCnt);

    k_init_acc<<<(NTOK * H_ / 4 + 255) / 256, 256, 0, stream>>>(x, acc, NTOK * H_ / 4);
    k_init_route<<<(E_ * CAP_ + 255) / 256, 256, 0, stream>>>(tokA, gateA, cnt);
    k_router<<<NTOK / 8, 256, 0, stream>>>(x, Wr, br, tokA, gateA, cnt);
    k_gather<<<E_ * CAP_, 256, 0, stream>>>(x, tokA, Xe);

    dim3 g1(F_ / 128, CAP_ / 128, E_);   // GEMM1: M=1280,N=4096,K=1024
    k_gemm<0><<<g1, 256, 0, stream>>>(Xe, W1, b1, H1, nullptr, nullptr, nullptr,
                                      CAP_, F_, H_);
    dim3 g2(H_ / 128, CAP_ / 128, E_);   // GEMM2: M=1280,N=1024,K=4096
    k_gemm<1><<<g2, 256, 0, stream>>>(H1, W2, b2, nullptr, tokA, gateA, acc,
                                      CAP_, H_, F_);
    k_layernorm<<<NTOK, 256, 0, stream>>>(acc, gamma, beta, out);
}
